// ChildSumTreeLSTM_67362267070628
// MI455X (gfx1250) — matrix operand order repaired and checked by reading, not compiled, not run on hardware
//
#include <hip/hip_runtime.h>
#include <hip/hip_bf16.h>
#include <math.h>

// ---------------------------------------------------------------------------
// Child-Sum Tree-LSTM for MI455X (gfx1250, wave32, WMMA bf16 + TDM).
//
//   * gather/matmul commute:  (h[idx]) @ W == (h @ W)[idx]
//     -> per level ONE fused GEMM  Hproj[8192,2048] = h_bf16 @ [Wih|Wfh|Woh|Wuh]
//   * input projections fused once (x zero-padded to K=512 so all GEMMs share
//     one shape and one TDM tile geometry).
//   * GEMM: 256-thread block = 8 waves, block tile 256x64. The 64x512 bf16
//     B strip (64KB) is staged once per block into LDS by the Tensor Data
//     Mover (tensor_load_to_lds, TENSORcnt) with a 16B/row pad so fragment
//     reads are LDS-bank-conflict-free. Each wave does 32x64 register tile,
//     8x v_wmma_f32_16x16x32_bf16 per K-step.
// ---------------------------------------------------------------------------

typedef __attribute__((ext_vector_type(16))) __bf16 v16bf;
typedef __attribute__((ext_vector_type(8)))  __bf16 v8bf;
typedef __attribute__((ext_vector_type(8)))  float  v8f;
typedef unsigned int u32;
typedef __attribute__((ext_vector_type(4))) u32 v4u;
typedef __attribute__((ext_vector_type(8))) int  v8i_;
typedef __attribute__((ext_vector_type(4))) int  v4i_;

#define NND   8192   // nodes
#define KCH   6      // children
#define NLVL  16     // levels (reference constant L)
#define INF   300    // embedding dim
#define HD    512    // hidden dim
#define KDIM  512    // shared GEMM K (x padded 300 -> 512)
#define NG    2048   // 4 gates * HD  (gate order: i, f, o, u)

#define BROWS 256    // M rows per block (8 waves x 32)
#define WROWS 32     // M rows per wave
#define BCOLS 64     // N cols per block
#define LPITCH 1040  // staged LDS row pitch: 1024B data + 16B pad (bank skew)

#if defined(__has_builtin)
#  if __has_builtin(__builtin_amdgcn_tensor_load_to_lds)
#    define HAVE_TDM 1
#  endif
#endif

__device__ __forceinline__ float sigmoidf_(float x) {
  return 1.0f / (1.0f + __expf(-x));
}

// ---- pack 4 weight matrices [Kreal, HD] fp32 -> transposed bf16 [NG, KDIM] ----
__global__ __launch_bounds__(256) void pack_w_kernel(
    const float* __restrict__ W0, const float* __restrict__ W1,
    const float* __restrict__ W2, const float* __restrict__ W3,
    __bf16* __restrict__ Wt, int Kreal) {
  int idx = blockIdx.x * 256 + threadIdx.x;   // over NG * KDIM
  int k   = idx & (KDIM - 1);
  int row = idx >> 9;                         // output column (gate*HD + n)
  if (row >= NG) return;
  int g  = row >> 9;
  int nn = row & (HD - 1);
  const float* W = (g == 0) ? W0 : (g == 1) ? W1 : (g == 2) ? W2 : W3;
  float v = (k < Kreal) ? W[(size_t)k * HD + nn] : 0.0f;
  Wt[(size_t)row * KDIM + k] = (__bf16)v;
}

// ---- embedding gather + fp32->bf16, zero-padded to KDIM ----
__global__ __launch_bounds__(256) void gather_x_kernel(
    const int* __restrict__ ids, const float* __restrict__ emb,
    __bf16* __restrict__ xbf) {
  int idx = blockIdx.x * 256 + threadIdx.x;   // over NND * KDIM
  int k = idx & (KDIM - 1);
  int n = idx >> 9;
  if (n >= NND) return;
  float v = (k < INF) ? emb[(size_t)ids[n] * INF + k] : 0.0f;
  xbf[(size_t)n * KDIM + k] = (__bf16)v;
}

// ---- bf16 WMMA GEMM: C[NND, NG] = A[NND, KDIM] * B,  B transposed Bt[NG, KDIM]
// Fragment layouts per CDNA5 ISA 7.12.2:
//   A 16x32 bf16 : lanes 0-15 -> M=lane,    K in {0..7, 16..23}
//                  lanes 16-31 -> M=lane-16, K in {8..15, 24..31}
//   B 32x16 bf16 : lanes 0-15 -> N=lane, K in {0..7, 16..23} ; lanes 16-31 -> K in {8..15, 24..31}
//   C/D 16x16 f32: vgpr v -> M = v + 8*(lane>=16), N = lane&15
__global__ __launch_bounds__(256) void wmma_gemm_bf16(
    const __bf16* __restrict__ A, const __bf16* __restrict__ Bt,
    float* __restrict__ C) {
  __shared__ __align__(128) char sB[BCOLS * LPITCH];   // 65 KB staged B strip

  const int tid  = threadIdx.x;
  const int wave = tid >> 5;
  const int lane = tid & 31;
  const int half = lane >> 4;
  const int l16  = lane & 15;
  const int n0   = blockIdx.y * BCOLS;
  const int m0   = blockIdx.x * BROWS + wave * WROWS;

#if defined(HAVE_TDM)
  // ---- Tensor Data Mover: stage Bt[n0..n0+63][0..511] -> LDS, 16B row pad ----
  if (wave == 0) {
    unsigned ldsa = (unsigned)(size_t)(&sB[0]);                    // LDS byte addr
    unsigned long long ga =
        (unsigned long long)(size_t)(Bt + (size_t)n0 * KDIM);      // tile start
    v4u g0;
    g0.x = 1u;                                        // count = 1 (valid D#)
    g0.y = ldsa;                                      // lds_addr
    g0.z = (u32)ga;                                   // global_addr[31:0]
    g0.w = (u32)((ga >> 32) & 0x01FFFFFFu)            // global_addr[56:32]
         | (2u << 30);                                // type = 2 (image)
    v8i_ g1;
    g1[0] = (int)((1u << 16)      // data_size = 1 -> 2-byte elements
                | (1u << 20)      // pad_enable
                | (7u << 22)      // pad_interval: every 1024B (one row)
                | (3u << 25));    // pad_amount: 4 DWORDs = 16B
    g1[1] = (int)((u32)(KDIM & 0xFFFF) << 16);              // tensor_dim0 lo16
    g1[2] = (int)(((u32)KDIM >> 16) | ((u32)BCOLS << 16));  // dim0 hi | dim1 lo
    g1[3] = (int)((u32)KDIM << 16);                         // dim1 hi | tile_dim0
    g1[4] = (int)BCOLS;                                     // tile_dim1 (rows)
    g1[5] = (int)KDIM;                                      // dim0_stride lo32
    g1[6] = 0;
    g1[7] = 0;
    v4i_ gz = {0, 0, 0, 0};
#if __clang_major__ >= 23
    v8i_ gz8 = {0, 0, 0, 0, 0, 0, 0, 0};
    __builtin_amdgcn_tensor_load_to_lds(g0, g1, gz, gz, gz8, 0);
#else
    __builtin_amdgcn_tensor_load_to_lds(g0, g1, gz, gz, 0);
#endif
    __builtin_amdgcn_s_wait_tensorcnt(0);
  }
#else
  // fallback: cooperative copy with the same padded layout
  for (int i = tid; i < BCOLS * (KDIM / 8); i += 256) {
    int r = i >> 6, c = i & 63;
    *(v8bf*)(sB + (size_t)r * LPITCH + c * 16) =
        *(const v8bf*)(Bt + ((size_t)n0 + r) * KDIM + c * 8);
  }
#endif
  __syncthreads();

  v8f acc[2][4] = {};
  const __bf16* Ar0 = A + (size_t)(m0 + l16) * KDIM;
  const __bf16* Ar1 = A + (size_t)(m0 + 16 + l16) * KDIM;

  for (int k0 = 0; k0 < KDIM; k0 += 32) {
    v8bf a0lo = *(const v8bf*)(Ar0 + k0 + half * 8);
    v8bf a0hi = *(const v8bf*)(Ar0 + k0 + 16 + half * 8);
    v8bf a1lo = *(const v8bf*)(Ar1 + k0 + half * 8);
    v8bf a1hi = *(const v8bf*)(Ar1 + k0 + 16 + half * 8);
    v16bf a0 = __builtin_shufflevector(a0lo, a0hi,
                 0,1,2,3,4,5,6,7,8,9,10,11,12,13,14,15);
    v16bf a1 = __builtin_shufflevector(a1lo, a1hi,
                 0,1,2,3,4,5,6,7,8,9,10,11,12,13,14,15);
#pragma unroll
    for (int nt = 0; nt < 4; nt++) {
      const char* bp = sB + (size_t)(nt * 16 + l16) * LPITCH
                          + (size_t)(k0 + half * 8) * 2;
      v8bf blo = *(const v8bf*)bp;
      v8bf bhi = *(const v8bf*)(bp + 32);
      v16bf b = __builtin_shufflevector(blo, bhi,
                 0,1,2,3,4,5,6,7,8,9,10,11,12,13,14,15);   // 2x ds_load_b128
      acc[0][nt] = __builtin_amdgcn_wmma_f32_16x16x32_bf16(
          false, a0, false, b, (short)0, acc[0][nt], false, false);
      acc[1][nt] = __builtin_amdgcn_wmma_f32_16x16x32_bf16(
          false, a1, false, b, (short)0, acc[1][nt], false, false);
    }
  }

#pragma unroll
  for (int mt = 0; mt < 2; mt++) {
    float* Cb = C + (size_t)(m0 + mt * 16 + half * 8) * NG + n0 + l16;
#pragma unroll
    for (int v = 0; v < 8; v++) {
      Cb[(size_t)v * NG +  0] = acc[mt][0][v];
      Cb[(size_t)v * NG + 16] = acc[mt][1][v];
      Cb[(size_t)v * NG + 32] = acc[mt][2][v];
      Cb[(size_t)v * NG + 48] = acc[mt][3][v];
    }
  }
}

// ---- per-level pointwise kernel: one block per node, thread j = hidden index ----
__global__ __launch_bounds__(HD) void treelstm_level_kernel(
    const float* __restrict__ Xproj, const float* __restrict__ Hproj,
    const int* __restrict__ child_idx, const int* __restrict__ child_mask,
    const int* __restrict__ node_level,
    const float* __restrict__ b_ix, const float* __restrict__ b_ih,
    const float* __restrict__ b_fx, const float* __restrict__ b_fh,
    const float* __restrict__ b_ox, const float* __restrict__ b_oh,
    const float* __restrict__ b_ux, const float* __restrict__ b_uh,
    const float* __restrict__ h_cur, const float* __restrict__ c_cur,
    float* __restrict__ h_nxt, float* __restrict__ c_nxt,
    __bf16* __restrict__ hbf_nxt, int lvl) {
  const int n = blockIdx.x;
  const int j = threadIdx.x;

  __shared__ int   sidx[KCH];
  __shared__ float smask[KCH];
  if (j < KCH) {
    sidx[j]  = child_idx[(size_t)n * KCH + j];
    smask[j] = (float)child_mask[(size_t)n * KCH + j];
  }
  __syncthreads();

  const float* xrow = Xproj + (size_t)n * NG;
  const float xf_pre = xrow[HD + j] + b_fx[j] + b_fh[j];   // gate f (block 1)

  float si = 0.f, so = 0.f, su = 0.f, fsum = 0.f;
#pragma unroll
  for (int k = 0; k < KCH; k++) {
    const float m  = smask[k];
    const int   ci = sidx[k];
    const float* hp = Hproj + (size_t)ci * NG;
    si   += m * hp[j];                                           // i proj
    fsum += m * sigmoidf_(xf_pre + hp[HD + j])                   // per-child f
              * c_cur[(size_t)ci * HD + j];
    so   += m * hp[2 * HD + j];                                  // o proj
    su   += m * hp[3 * HD + j];                                  // u proj
  }

  const float i = sigmoidf_(xrow[j]          + b_ix[j] + b_ih[j] + si);
  const float o = sigmoidf_(xrow[2 * HD + j] + b_ox[j] + b_oh[j] + so);
  const float u = tanhf(    xrow[3 * HD + j] + b_ux[j] + b_uh[j] + su);
  const float cn = i * u + fsum;
  const float hn = o * tanhf(cn);

  const bool upd = (node_level[n] == lvl);
  const size_t off = (size_t)n * HD + j;
  const float co = upd ? cn : c_cur[off];
  const float ho = upd ? hn : h_cur[off];
  c_nxt[off]   = co;
  h_nxt[off]   = ho;
  hbf_nxt[off] = (__bf16)ho;
}

// ---------------------------------------------------------------------------
extern "C" void kernel_launch(void* const* d_in, const int* in_sizes, int n_in,
                              void* d_out, int out_size, void* d_ws, size_t ws_size,
                              hipStream_t stream) {
  (void)in_sizes; (void)n_in; (void)out_size; (void)ws_size;

  const int*   input_ids  = (const int*)d_in[0];
  const int*   child_idx  = (const int*)d_in[1];
  const int*   child_mask = (const int*)d_in[2];
  const int*   node_level = (const int*)d_in[3];
  /* d_in[4] = num_levels scalar; reference constant L = 16 (hardcoded) */
  const float* emb  = (const float*)d_in[5];
  const float* W_ix = (const float*)d_in[6];  const float* b_ix = (const float*)d_in[7];
  const float* W_ih = (const float*)d_in[8];  const float* b_ih = (const float*)d_in[9];
  const float* W_fx = (const float*)d_in[10]; const float* b_fx = (const float*)d_in[11];
  const float* W_fh = (const float*)d_in[12]; const float* b_fh = (const float*)d_in[13];
  const float* W_ox = (const float*)d_in[14]; const float* b_ox = (const float*)d_in[15];
  const float* W_oh = (const float*)d_in[16]; const float* b_oh = (const float*)d_in[17];
  const float* W_ux = (const float*)d_in[18]; const float* b_ux = (const float*)d_in[19];
  const float* W_uh = (const float*)d_in[20]; const float* b_uh = (const float*)d_in[21];

  // ---- workspace carve-out (256B aligned regions) ----
  char* ws = (char*)d_ws;
  size_t off = 0;
  auto carve = [&](size_t bytes) -> char* {
    char* p = ws + off;
    off += (bytes + 255) & ~(size_t)255;
    return p;
  };
  float*  Xproj = (float*)carve((size_t)NND * NG * 4);
  float*  Hproj = (float*)carve((size_t)NND * NG * 4);
  float*  hA    = (float*)carve((size_t)NND * HD * 4);
  float*  hB    = (float*)carve((size_t)NND * HD * 4);
  float*  cA    = (float*)carve((size_t)NND * HD * 4);
  float*  cB    = (float*)carve((size_t)NND * HD * 4);
  __bf16* hbfA  = (__bf16*)carve((size_t)NND * HD * 2);
  __bf16* hbfB  = (__bf16*)carve((size_t)NND * HD * 2);
  __bf16* xbf   = (__bf16*)carve((size_t)NND * KDIM * 2);
  __bf16* WxT   = (__bf16*)carve((size_t)NG * KDIM * 2);
  __bf16* WhT   = (__bf16*)carve((size_t)NG * KDIM * 2);

  // ---- zero the level-0 state (graph-capture-safe memsets) ----
  hipMemsetAsync(hA,   0, (size_t)NND * HD * 4, stream);
  hipMemsetAsync(cA,   0, (size_t)NND * HD * 4, stream);
  hipMemsetAsync(hbfA, 0, (size_t)NND * HD * 2, stream);

  // ---- one-time packing (gate order i, f, o, u) ----
  pack_w_kernel<<<(NG * KDIM + 255) / 256, 256, 0, stream>>>(
      W_ix, W_fx, W_ox, W_ux, WxT, INF);
  pack_w_kernel<<<(NG * KDIM + 255) / 256, 256, 0, stream>>>(
      W_ih, W_fh, W_oh, W_uh, WhT, HD);
  gather_x_kernel<<<(NND * KDIM + 255) / 256, 256, 0, stream>>>(
      input_ids, emb, xbf);

  // ---- input projections: Xproj = x_bf16 @ [Wix|Wfx|Wox|Wux] ----
  dim3 ggrid(NND / BROWS, NG / BCOLS);
  wmma_gemm_bf16<<<ggrid, 256, 0, stream>>>(xbf, WxT, Xproj);

  // ---- level loop with h/c ping-pong ----
  float*  h_cur = hA;  float*  h_nxt = hB;
  float*  c_cur = cA;  float*  c_nxt = cB;
  __bf16* hbf_cur = hbfA; __bf16* hbf_nxt = hbfB;

  for (int lvl = 0; lvl < NLVL; ++lvl) {
    // Hproj = h_bf16 @ [Wih|Wfh|Woh|Wuh]
    wmma_gemm_bf16<<<ggrid, 256, 0, stream>>>(hbf_cur, WhT, Hproj);

    treelstm_level_kernel<<<NND, HD, 0, stream>>>(
        Xproj, Hproj, child_idx, child_mask, node_level,
        b_ix, b_ih, b_fx, b_fh, b_ox, b_oh, b_ux, b_uh,
        h_cur, c_cur, h_nxt, c_nxt, hbf_nxt, lvl);

    { float* t = h_cur; h_cur = h_nxt; h_nxt = t; }
    { float* t = c_cur; c_cur = c_nxt; c_nxt = t; }
    { __bf16* t = hbf_cur; hbf_cur = hbf_nxt; hbf_nxt = t; }
  }

  // ---- final hidden states -> d_out (fp32 [N, H]) ----
  hipMemcpyAsync(d_out, h_cur, (size_t)NND * HD * 4,
                 hipMemcpyDeviceToDevice, stream);
}
